// Decoder_78460462564101
// MI455X (gfx1250) — matrix operand order repaired and checked by reading, not compiled, not run on hardware
//
#include <hip/hip_runtime.h>

// ---------------------------------------------------------------------------
// LSTM caption decoder for MI455X (gfx1250), wave32 + WMMA bf16 16x16x32.
// GEMMs use v_wmma_f32_16x16x32_bf16 with f32 accumulation, biases folded
// into C init, and CDNA5 async global->LDS staging of the shared A tile
// (global_load_async_to_lds_b128 + s_wait_asynccnt + barrier + ds_load).
// ---------------------------------------------------------------------------

typedef __attribute__((ext_vector_type(16))) __bf16 v16bf;
typedef __attribute__((ext_vector_type(8)))  __bf16 v8bf;
typedef __attribute__((ext_vector_type(8)))  float  v8f;

#define Bz  128
#define Tz  64
#define Ez  512
#define Hz  1024
#define Vz  10000
#define G4H 4096   // 4*H

// float -> bf16, round-to-nearest-even
static __device__ __forceinline__ unsigned short f2bf(float x) {
  union { float f; unsigned int u; } v; v.f = x;
  unsigned int u = v.u;
  u += 0x7FFFu + ((u >> 16) & 1u);
  return (unsigned short)(u >> 16);
}

static __device__ __forceinline__ float sigf(float x) {
  return 1.0f / (1.0f + __expf(-x));
}

// A fragment: 16x32 bf16 (M x K). Lane L: m = L&15, khalf = L>>4.
// elems 0..7  = A[m, khalf*8 + 0..7]   (VGPR0..3)
// elems 8..15 = A[m, 16 + khalf*8 + 0..7] (VGPR4..7)
static __device__ __forceinline__ v16bf load_a_frag(const __bf16* p,
                                                    int lda, int lane) {
  const int m  = lane & 15;
  const int kh = lane >> 4;
  const __bf16* q = p + m * lda + kh * 8;
  v8bf lo = *(const v8bf*)(q);
  v8bf hi = *(const v8bf*)(q + 16);
  return __builtin_shufflevector(lo, hi, 0,1,2,3,4,5,6,7,8,9,10,11,12,13,14,15);
}

// B fragment: 32x16 bf16 (K x N), B stored row-major as [N, K] (TN gemm).
// Lane L: n = L&15, kq = L>>4; elems 0..7 = B[n, kq*8 + 0..7], 8..15 = B[n, 16 + kq*8 + 0..7].
static __device__ __forceinline__ v16bf load_b_frag(const __bf16* __restrict__ p,
                                                    int ldb, int lane) {
  const int n  = lane & 15;
  const int kq = lane >> 4;
  const __bf16* q = p + n * ldb + kq * 8;
  v8bf lo = *(const v8bf*)(q);
  v8bf hi = *(const v8bf*)(q + 16);
  return __builtin_shufflevector(lo, hi, 0,1,2,3,4,5,6,7,8,9,10,11,12,13,14,15);
}

static __device__ __forceinline__ v8f wmma_bf16(v16bf a, v16bf b, v8f c) {
  return __builtin_amdgcn_wmma_f32_16x16x32_bf16(
      /*neg_a=*/false, a, /*neg_b=*/false, b,
      /*c_mod=*/(short)0, c, /*reuse_a=*/false, /*reuse_b=*/false);
}

// ---------------------------------------------------------------------------
// Async-stage a contiguous 16 x K bf16 A-tile (rows consecutive in memory)
// from global into LDS using the CDNA5 Tensor/async path:
//   global_load_async_to_lds_b128 (ASYNCcnt) -> s_wait_asynccnt 0 -> barrier.
// 256 threads, 16 bytes per thread per chunk; K/128 chunks of 4 KB.
// ---------------------------------------------------------------------------
template <int K>
static __device__ __forceinline__ void stage_a_async(const void* gsrc, void* lds) {
  const unsigned           lbase = (unsigned)(unsigned long long)(uintptr_t)lds;
  const unsigned long long gbase = (unsigned long long)(uintptr_t)gsrc;
  constexpr int chunks = (16 * K * 2) / (256 * 16);   // 4KB per chunk
#pragma unroll
  for (int j = 0; j < chunks; ++j) {
    const unsigned           loff = lbase + threadIdx.x * 16u + (unsigned)j * 4096u;
    const unsigned long long goff = gbase + threadIdx.x * 16u + (unsigned)j * 4096u;
    asm volatile("global_load_async_to_lds_b128 %0, %1, off"
                 :: "v"(loff), "v"(goff) : "memory");
  }
  asm volatile("s_wait_asynccnt 0x0" ::: "memory");
  __syncthreads();
}

// ---------------------------------------------------------------------------
// Elementwise f32 -> bf16 weight conversion
// ---------------------------------------------------------------------------
__global__ void k_convert_bf16(const float* __restrict__ src,
                               unsigned short* __restrict__ dst, int n) {
  int i = blockIdx.x * blockDim.x + threadIdx.x;
  if (i < n) dst[i] = f2bf(src[i]);
}

// ---------------------------------------------------------------------------
// Build xs[T*B, E] in bf16: row r = t*B + b; t==0 -> features[b], else
// embed_W[captions[b, t]].  128 threads/row, 4 elems/thread (E=512).
// ---------------------------------------------------------------------------
__global__ __launch_bounds__(128)
void k_build_xs(const float* __restrict__ features,
                const int*   __restrict__ captions,
                const float* __restrict__ embed_W,
                unsigned short* __restrict__ xs) {
  const int r = blockIdx.x;          // 0..T*B-1, r = t*B + b
  const int t = r >> 7;              // B = 128
  const int b = r & 127;
  const float* src;
  if (t == 0) {
    src = features + (size_t)b * Ez;
  } else {
    int idx = captions[b * Tz + t];
    src = embed_W + (size_t)idx * Ez;
  }
  unsigned short* dst = xs + (size_t)r * Ez;
  const int e = threadIdx.x * 4;
  float4 v = *(const float4*)(src + e);
  dst[e + 0] = f2bf(v.x);
  dst[e + 1] = f2bf(v.y);
  dst[e + 2] = f2bf(v.z);
  dst[e + 3] = f2bf(v.w);
}

// ---------------------------------------------------------------------------
// GEMM 1: xg[T*B, 4H] = Xs[T*B, E] @ W_ih[4H, E]^T + (b_ih + b_hh)
// A tile (16 x 512, 16KB) async-staged to LDS, shared by 8 waves.
// Wave computes 16x32; block tile M=16, N=256; grid = (16, 512).
// ---------------------------------------------------------------------------
__global__ __launch_bounds__(256)
void k_gemm_xgates(const __bf16* __restrict__ A,   // [8192, 512]
                   const __bf16* __restrict__ Bw,  // [4096, 512]
                   const float*  __restrict__ b_ih,
                   const float*  __restrict__ b_hh,
                   float* __restrict__ C) {        // [8192, 4096]
  __shared__ __align__(16) unsigned short sA[16 * Ez];   // 16 KB
  const int lane = threadIdx.x & 31;
  const int wave = threadIdx.x >> 5;
  const int tile_m = blockIdx.y * 16;
  const int n0 = blockIdx.x * 256 + wave * 32;
  const int n  = lane & 15;

  stage_a_async<Ez>((const void*)(A + (size_t)tile_m * Ez), (void*)sA);

  v8f c0, c1;
  const float bias0 = b_ih[n0 + n]      + b_hh[n0 + n];
  const float bias1 = b_ih[n0 + 16 + n] + b_hh[n0 + 16 + n];
#pragma unroll
  for (int v = 0; v < 8; ++v) { c0[v] = bias0; c1[v] = bias1; }

  const __bf16* As = (const __bf16*)sA;
  const __bf16* B0 = Bw + (size_t)n0 * Ez;
  const __bf16* B1 = Bw + (size_t)(n0 + 16) * Ez;

  for (int k = 0; k < Ez; k += 32) {
    __builtin_prefetch((const void*)(B0 + k + 128), 0, 1);
    v16bf a  = load_a_frag(As + k, Ez, lane);     // ds_load from LDS
    v16bf b0 = load_b_frag(B0 + k, Ez, lane);
    v16bf b1 = load_b_frag(B1 + k, Ez, lane);
    c0 = wmma_bf16(a, b0, c0);
    c1 = wmma_bf16(a, b1, c1);
  }

  const int row0 = (lane >> 4) * 8;
#pragma unroll
  for (int v = 0; v < 8; ++v) {
    const int r = tile_m + row0 + v;
    C[(size_t)r * G4H + n0 + n]      = c0[v];
    C[(size_t)r * G4H + n0 + 16 + n] = c1[v];
  }
}

// ---------------------------------------------------------------------------
// GEMM 2 (per time step): gbuf[B, 4H] = xg_t + h[B, H] @ W_hh[4H, H]^T
// A tile (16 x 1024, 32KB) async-staged to LDS. Wave 16x32; grid = (16, 8).
// ---------------------------------------------------------------------------
__global__ __launch_bounds__(256)
void k_gemm_hh(const __bf16* __restrict__ A,    // h bf16 [128, 1024]
               const __bf16* __restrict__ Bw,   // W_hh bf16 [4096, 1024]
               const float*  __restrict__ Cinit,// xg_t [128, 4096]
               float* __restrict__ C) {         // gbuf [128, 4096]
  __shared__ __align__(16) unsigned short sA[16 * Hz];   // 32 KB
  const int lane = threadIdx.x & 31;
  const int wave = threadIdx.x >> 5;
  const int tile_m = blockIdx.y * 16;
  const int n0 = blockIdx.x * 256 + wave * 32;
  const int n  = lane & 15;
  const int row0 = (lane >> 4) * 8;

  stage_a_async<Hz>((const void*)(A + (size_t)tile_m * Hz), (void*)sA);

  v8f c0, c1;
#pragma unroll
  for (int v = 0; v < 8; ++v) {
    const int r = tile_m + row0 + v;
    c0[v] = Cinit[(size_t)r * G4H + n0 + n];
    c1[v] = Cinit[(size_t)r * G4H + n0 + 16 + n];
  }

  const __bf16* As = (const __bf16*)sA;
  const __bf16* B0 = Bw + (size_t)n0 * Hz;
  const __bf16* B1 = Bw + (size_t)(n0 + 16) * Hz;

  for (int k = 0; k < Hz; k += 32) {
    __builtin_prefetch((const void*)(B0 + k + 128), 0, 1);
    v16bf a  = load_a_frag(As + k, Hz, lane);
    v16bf b0 = load_b_frag(B0 + k, Hz, lane);
    v16bf b1 = load_b_frag(B1 + k, Hz, lane);
    c0 = wmma_bf16(a, b0, c0);
    c1 = wmma_bf16(a, b1, c1);
  }

#pragma unroll
  for (int v = 0; v < 8; ++v) {
    const int r = tile_m + row0 + v;
    C[(size_t)r * G4H + n0 + n]      = c0[v];
    C[(size_t)r * G4H + n0 + 16 + n] = c1[v];
  }
}

// ---------------------------------------------------------------------------
// LSTM pointwise: gates -> c, h (h written as bf16 to state + history)
// ---------------------------------------------------------------------------
__global__ __launch_bounds__(256)
void k_lstm_pointwise(const float* __restrict__ g,     // [128, 4096]
                      float* __restrict__ c,            // [128, 1024]
                      unsigned short* __restrict__ h,   // [128, 1024] bf16
                      unsigned short* __restrict__ hs_t)// [128, 1024] bf16
{
  const int idx = blockIdx.x * blockDim.x + threadIdx.x;
  const int b = idx >> 10;
  const int j = idx & 1023;
  const float* gb = g + (size_t)b * G4H;
  const float i_g = sigf(gb[j]);
  const float f_g = sigf(gb[Hz + j]);
  const float g_g = tanhf(gb[2 * Hz + j]);
  const float o_g = sigf(gb[3 * Hz + j]);
  const float cn = f_g * c[idx] + i_g * g_g;
  c[idx] = cn;
  const unsigned short hb = f2bf(o_g * tanhf(cn));
  h[idx] = hb;
  hs_t[idx] = hb;
}

// ---------------------------------------------------------------------------
// GEMM 3: out[b, t, v] = hs[t*B+b, H] @ fc_W[V, H]^T + fc_b
// Dominant GEMM: A tile async-staged to LDS; wave owns 4 N-tiles (16x64).
// N = 10000 = 625 tiles of 16; 157 quads, wave-uniform clamps.
// grid = (20, 512), block = 256 (8 waves x 4 tiles = 32 tiles/block).
// ---------------------------------------------------------------------------
__global__ __launch_bounds__(256)
void k_gemm_fc(const __bf16* __restrict__ A,    // hs bf16 [8192, 1024]
               const __bf16* __restrict__ Bw,   // fc_W bf16 [10000, 1024]
               const float*  __restrict__ bias, // fc_b [10000]
               float* __restrict__ out) {       // [128, 64, 10000]
  __shared__ __align__(16) unsigned short sA[16 * Hz];   // 32 KB
  const int lane = threadIdx.x & 31;
  const int wave = threadIdx.x >> 5;
  const int tile_m = blockIdx.y * 16;
  const int q0 = (blockIdx.x * 8 + wave) * 4;   // first of 4 N-tiles
  const int n = lane & 15;
  const int row0 = (lane >> 4) * 8;

  stage_a_async<Hz>((const void*)(A + (size_t)tile_m * Hz), (void*)sA);

  int  tnc[4];
  bool ok[4];
  const __bf16* Bp[4];
  v8f acc[4];
#pragma unroll
  for (int i = 0; i < 4; ++i) {
    const int tn = q0 + i;
    ok[i]  = tn < 625;                 // wave-uniform guard
    tnc[i] = ok[i] ? tn : 624;         // clamp keeps loads in range
    Bp[i]  = Bw + (size_t)tnc[i] * 16 * Hz;
    const float bv = bias[tnc[i] * 16 + n];
#pragma unroll
    for (int v = 0; v < 8; ++v) acc[i][v] = bv;
  }

  const __bf16* As = (const __bf16*)sA;
  for (int k = 0; k < Hz; k += 32) {
    __builtin_prefetch((const void*)(Bp[0] + k + 128), 0, 1);
    __builtin_prefetch((const void*)(Bp[2] + k + 128), 0, 1);
    v16bf a = load_a_frag(As + k, Hz, lane);
#pragma unroll
    for (int i = 0; i < 4; ++i) {
      v16bf bf = load_b_frag(Bp[i] + k, Hz, lane);
      acc[i] = wmma_bf16(a, bf, acc[i]);
    }
  }

#pragma unroll
  for (int v = 0; v < 8; ++v) {
    const int r = tile_m + row0 + v;   // r = t*B + b
    const int t = r >> 7;
    const int b = r & 127;
    const size_t base = (size_t)b * (Tz * Vz) + (size_t)t * Vz;
#pragma unroll
    for (int i = 0; i < 4; ++i) {
      if (ok[i]) out[base + tnc[i] * 16 + n] = acc[i][v];
    }
  }
}

// ---------------------------------------------------------------------------
// Host launcher
// ---------------------------------------------------------------------------
extern "C" void kernel_launch(void* const* d_in, const int* in_sizes, int n_in,
                              void* d_out, int out_size, void* d_ws, size_t ws_size,
                              hipStream_t stream) {
  (void)in_sizes; (void)n_in; (void)out_size; (void)ws_size;

  const float* features = (const float*)d_in[0];
  const int*   captions = (const int*)  d_in[1];
  const float* embed_W  = (const float*)d_in[2];
  const float* W_ih     = (const float*)d_in[3];
  const float* W_hh     = (const float*)d_in[4];
  const float* b_ih     = (const float*)d_in[5];
  const float* b_hh     = (const float*)d_in[6];
  const float* fc_W     = (const float*)d_in[7];
  const float* fc_b     = (const float*)d_in[8];
  float* out = (float*)d_out;

  // ---- workspace carve-up (all 256B aligned) ----
  unsigned char* wsb = (unsigned char*)d_ws;
  size_t off = 0;
  auto carve = [&](size_t bytes) -> void* {
    void* p = wsb + off;
    off += (bytes + 255) & ~(size_t)255;
    return p;
  };
  float*          xg      = (float*)         carve((size_t)Tz * Bz * G4H * 4); // 128 MB
  unsigned short* xs_bf   = (unsigned short*)carve((size_t)Tz * Bz * Ez * 2);  //   8 MB
  unsigned short* Wih_bf  = (unsigned short*)carve((size_t)G4H * Ez * 2);      //   4 MB
  unsigned short* Whh_bf  = (unsigned short*)carve((size_t)G4H * Hz * 2);      //   8 MB
  unsigned short* fcW_bf  = (unsigned short*)carve((size_t)Vz * Hz * 2);       //  20 MB
  unsigned short* hs_bf   = (unsigned short*)carve((size_t)Tz * Bz * Hz * 2);  //  16 MB
  float*          gbuf    = (float*)         carve((size_t)Bz * G4H * 4);      //   2 MB
  float*          c_buf   = (float*)         carve((size_t)Bz * Hz * 4);       // 0.5 MB
  unsigned short* h_bf    = (unsigned short*)carve((size_t)Bz * Hz * 2);       // .25 MB

  // ---- state init (deterministic per call) ----
  hipMemsetAsync(c_buf, 0, (size_t)Bz * Hz * 4, stream);
  hipMemsetAsync(h_bf,  0, (size_t)Bz * Hz * 2, stream);

  // ---- weight conversions to bf16 ----
  {
    int n1 = G4H * Ez;  // 2,097,152
    int n2 = G4H * Hz;  // 4,194,304
    int n3 = Vz * Hz;   // 10,240,000
    k_convert_bf16<<<dim3((n1 + 255) / 256), dim3(256), 0, stream>>>(W_ih, Wih_bf, n1);
    k_convert_bf16<<<dim3((n2 + 255) / 256), dim3(256), 0, stream>>>(W_hh, Whh_bf, n2);
    k_convert_bf16<<<dim3((n3 + 255) / 256), dim3(256), 0, stream>>>(fc_W, fcW_bf, n3);
  }

  // ---- build input sequence (embedding gather + features at t=0) ----
  k_build_xs<<<dim3(Tz * Bz), dim3(128), 0, stream>>>(features, captions, embed_W, xs_bf);

  // ---- phase 1: gate precompute GEMM ----
  k_gemm_xgates<<<dim3(G4H / 256, (Tz * Bz) / 16), dim3(256), 0, stream>>>(
      (const __bf16*)xs_bf, (const __bf16*)Wih_bf, b_ih, b_hh, xg);

  // ---- phase 2: LSTM recurrence (graph-captured launch chain) ----
  for (int t = 0; t < Tz; ++t) {
    k_gemm_hh<<<dim3(G4H / 256, Bz / 16), dim3(256), 0, stream>>>(
        (const __bf16*)h_bf, (const __bf16*)Whh_bf,
        xg + (size_t)t * Bz * G4H, gbuf);
    k_lstm_pointwise<<<dim3((Bz * Hz) / 256), dim3(256), 0, stream>>>(
        gbuf, c_buf, h_bf, hs_bf + (size_t)t * Bz * Hz);
  }

  // ---- phase 3: output projection GEMM ----
  k_gemm_fc<<<dim3(20, (Tz * Bz) / 16), dim3(256), 0, stream>>>(
      (const __bf16*)hs_bf, (const __bf16*)fcW_bf, fc_b, out);
}
